// LSM_77129022702036
// MI455X (gfx1250) — matrix operand order repaired and checked by reading, not compiled, not run on hardware
//
#include <hip/hip_runtime.h>

// ---------------------------------------------------------------------------
// Problem sizes (fixed by the reference)
// ---------------------------------------------------------------------------
#define T_STEPS 1000
#define BATCH   64
#define IN_SZ   700
#define NRES    1000

#define K_IN_PAD 704    // IN_SZ padded to multiple of 32
#define K_H_PAD  1024   // NRES padded to multiple of 32 (K of recurrent GEMM)
#define N_PAD    1024   // NRES padded to multiple of 64 (N tiles)

#define NWG_STEP 16     // persistent workgroups in the recurrent kernel

typedef __attribute__((ext_vector_type(8)))  _Float16 f16x8;
typedef __attribute__((ext_vector_type(16))) _Float16 f16x16;
typedef __attribute__((ext_vector_type(8)))  float    f32x8;

// ---------------------------------------------------------------------------
// WMMA fragment loaders (layouts per CDNA5 ISA §7.12.2, wave32)
// ---------------------------------------------------------------------------
// A (16x32 f16, row-major, stride lda elems):
//   lane<16 : row=lane,    v0..3 = K[k0..k0+7],  v4..7 = K[k0+16..k0+23]
//   lane>=16: row=lane-16, K offsets +8
__device__ __forceinline__ f16x16 load_a_frag(const _Float16* base, int lda,
                                              int row0, int k0, int lane) {
    int half = lane >> 4;
    int r    = lane & 15;
    const _Float16* p = base + (size_t)(row0 + r) * lda + (k0 + half * 8);
    f16x16 a;
    ((f16x8*)&a)[0] = *(const f16x8*)(p);
    ((f16x8*)&a)[1] = *(const f16x8*)(p + 16);
    return a;
}

// B (32x16 f16) loaded from COLUMN-MAJOR storage (BT[n][k] contiguous in k,
// stride ldb elems): lane<16: col=lane, K=k0..k0+7,k0+16..k0+23 ; lane>=16: K offsets +8
__device__ __forceinline__ f16x16 load_b_frag(const _Float16* baseT, int ldb,
                                              int k0, int col0, int lane) {
    int half = lane >> 4;
    int c    = lane & 15;
    const _Float16* p = baseT + (size_t)(col0 + c) * ldb + (k0 + half * 8);
    f16x16 b;
    ((f16x8*)&b)[0] = *(const f16x8*)(p);
    ((f16x8*)&b)[1] = *(const f16x8*)(p + 16);
    return b;
}

__device__ __forceinline__ f32x8 wmma_f16(f16x16 a, f16x16 b, f32x8 c) {
    return __builtin_amdgcn_wmma_f32_16x16x32_f16(false, a, false, b,
                                                  (short)0, c, false, false);
}

// tanh via hardware v_exp_f32; saturation-safe at both ends.
__device__ __forceinline__ float fast_tanh(float x) {
    float e = __builtin_amdgcn_exp2f(x * 2.885390081777926814f); // exp(2x)
    return 1.0f - 2.0f / (e + 1.0f);
}

// ---------------------------------------------------------------------------
// Prep kernels: fp16 conversion with zero padding, buffer/barrier init
// ---------------------------------------------------------------------------
__global__ void __launch_bounds__(256)
init_kernel(unsigned* bar, _Float16* h0) {
    if (blockIdx.x == 0 && threadIdx.x == 0) { bar[0] = 0u; bar[1] = 0u; }
    size_t stride = (size_t)gridDim.x * blockDim.x;
    for (size_t i = (size_t)blockIdx.x * blockDim.x + threadIdx.x;
         i < (size_t)BATCH * K_H_PAD; i += stride)
        h0[i] = (_Float16)0.0f;
}

__global__ void __launch_bounds__(256)
cvt_x_kernel(const float* __restrict__ x, _Float16* __restrict__ xb) {
    const size_t total = (size_t)T_STEPS * BATCH * K_IN_PAD;
    size_t stride = (size_t)gridDim.x * blockDim.x;
    for (size_t i = (size_t)blockIdx.x * blockDim.x + threadIdx.x;
         i < total; i += stride) {
        size_t row = i / K_IN_PAD;
        int    col = (int)(i - row * K_IN_PAD);
        float  v   = (col < IN_SZ) ? x[row * IN_SZ + col] : 0.0f;
        xb[i] = (_Float16)v;
    }
}

__global__ void __launch_bounds__(256)
cvt_win_kernel(const float* __restrict__ win, _Float16* __restrict__ winb) {
    const size_t total = (size_t)N_PAD * K_IN_PAD;   // winb[n][k]
    size_t stride = (size_t)gridDim.x * blockDim.x;
    for (size_t i = (size_t)blockIdx.x * blockDim.x + threadIdx.x;
         i < total; i += stride) {
        int n = (int)(i / K_IN_PAD);
        int k = (int)(i - (size_t)n * K_IN_PAD);
        float v = (n < NRES && k < IN_SZ) ? win[(size_t)n * IN_SZ + k] : 0.0f;
        winb[i] = (_Float16)v;
    }
}

__global__ void __launch_bounds__(256)
cvt_wlsm_kernel(const float* __restrict__ wlsm, _Float16* __restrict__ wt) {
    // wt[n][k] = Wlsm[k][n]  (transpose so B columns are contiguous in K)
    const size_t total = (size_t)N_PAD * K_H_PAD;
    size_t stride = (size_t)gridDim.x * blockDim.x;
    for (size_t i = (size_t)blockIdx.x * blockDim.x + threadIdx.x;
         i < total; i += stride) {
        int n = (int)(i >> 10);        // / K_H_PAD
        int k = (int)(i & (K_H_PAD - 1));
        float v = (n < NRES && k < NRES) ? wlsm[(size_t)k * NRES + n] : 0.0f;
        wt[i] = (_Float16)v;
    }
}

// ---------------------------------------------------------------------------
// Stage A: pre[t,b,n] = x[t,b,:] @ Win[n,:] + bias[n]  -> written into d_out
// M = T*B = 64000, K = 704, N = 1024 (store only n < 1000)
// Block = 256 thr (8 waves) computes a 64x64 tile; each wave 16x32 of output.
// ---------------------------------------------------------------------------
__global__ void __launch_bounds__(256)
gemm_pre_kernel(const _Float16* __restrict__ xb, const _Float16* __restrict__ winb,
                const float* __restrict__ bias, float* __restrict__ out) {
    int lane = threadIdx.x & 31;
    int wave = threadIdx.x >> 5;
    int row0 = blockIdx.y * 64 + (wave & 3) * 16;
    int col0 = blockIdx.x * 64 + (wave >> 2) * 32;

    f32x8 acc0 = {}; f32x8 acc1 = {};
    for (int k0 = 0; k0 < K_IN_PAD; k0 += 32) {
        f16x16 a  = load_a_frag(xb,   K_IN_PAD, row0, k0, lane);
        f16x16 b0 = load_b_frag(winb, K_IN_PAD, k0, col0,      lane);
        f16x16 b1 = load_b_frag(winb, K_IN_PAD, k0, col0 + 16, lane);
        acc0 = wmma_f16(a, b0, acc0);
        acc1 = wmma_f16(a, b1, acc1);
    }

    int mlo = (lane >> 4) * 8;
    int nn  = lane & 15;
    for (int r = 0; r < 8; ++r) {
        int m  = row0 + mlo + r;
        int n0 = col0 + nn;
        int n1 = n0 + 16;
        if (n0 < NRES) out[(size_t)m * NRES + n0] = acc0[r] + bias[n0];
        if (n1 < NRES) out[(size_t)m * NRES + n1] = acc1[r] + bias[n1];
    }
}

// ---------------------------------------------------------------------------
// Grid-wide sense-reversing barrier over NWG_STEP resident workgroups
// ---------------------------------------------------------------------------
__device__ __forceinline__ void grid_barrier(unsigned* bar, unsigned* phase) {
    __threadfence();          // release our h / out stores device-wide
    __syncthreads();
    if (threadIdx.x == 0) {
        unsigned ph = *phase;
        unsigned old = __hip_atomic_fetch_add(&bar[0], 1u, __ATOMIC_ACQ_REL,
                                              __HIP_MEMORY_SCOPE_AGENT);
        if (old == (unsigned)(NWG_STEP - 1)) {
            __hip_atomic_store(&bar[0], 0u, __ATOMIC_RELAXED,
                               __HIP_MEMORY_SCOPE_AGENT);
            __hip_atomic_fetch_add(&bar[1], 1u, __ATOMIC_RELEASE,
                                   __HIP_MEMORY_SCOPE_AGENT);
        } else {
            while (__hip_atomic_load(&bar[1], __ATOMIC_ACQUIRE,
                                     __HIP_MEMORY_SCOPE_AGENT) == ph)
                __builtin_amdgcn_s_sleep(2);
        }
        *phase = ph + 1;
    }
    __syncthreads();
    __threadfence();          // acquire other WGs' h stores (invalidate WGP$)
}

// ---------------------------------------------------------------------------
// Stage B (persistent): for t in 0..T-1:
//   h_t = tanh(pre_t + h_{t-1} @ Wlsm) ; pre_t read from & h_t written to d_out
//   fp16 copy of h ping-pongs between h0/h1 (padded cols kept zero)
// 16 WGs x 256 thr; each WG owns 64 output columns; each wave 16x32 of h.
// ---------------------------------------------------------------------------
__global__ void __launch_bounds__(256)
reservoir_kernel(const _Float16* __restrict__ wlsmT, float* __restrict__ out,
                 _Float16* __restrict__ hbuf0, _Float16* __restrict__ hbuf1,
                 unsigned* bar) {
    int lane = threadIdx.x & 31;
    int wave = threadIdx.x >> 5;
    int row0 = (wave & 3) * 16;                    // BATCH=64 -> 4 M tiles
    int col0 = blockIdx.x * 64 + (wave >> 2) * 32; // 16 WGs cover N_PAD=1024
    int mlo  = (lane >> 4) * 8;
    int nn   = lane & 15;
    unsigned phase = 0;

    for (int t = 0; t < T_STEPS; ++t) {
        const _Float16* hprev = (t & 1) ? hbuf1 : hbuf0;
        _Float16*       hnext = (t & 1) ? hbuf0 : hbuf1;

        f32x8 acc0 = {}; f32x8 acc1 = {};
        #pragma unroll 4
        for (int k0 = 0; k0 < K_H_PAD; k0 += 32) {
            f16x16 a  = load_a_frag(hprev, K_H_PAD, row0, k0, lane);
            f16x16 b0 = load_b_frag(wlsmT, K_H_PAD, k0, col0,      lane);
            f16x16 b1 = load_b_frag(wlsmT, K_H_PAD, k0, col0 + 16, lane);
            acc0 = wmma_f16(a, b0, acc0);
            acc1 = wmma_f16(a, b1, acc1);
        }

        float* outt = out + (size_t)t * BATCH * NRES;
        for (int r = 0; r < 8; ++r) {
            int m  = row0 + mlo + r;
            int n0 = col0 + nn;
            int n1 = n0 + 16;
            float v0 = 0.0f, v1 = 0.0f;
            if (n0 < NRES) {
                v0 = fast_tanh(outt[(size_t)m * NRES + n0] + acc0[r]);
                outt[(size_t)m * NRES + n0] = v0;
            }
            if (n1 < NRES) {
                v1 = fast_tanh(outt[(size_t)m * NRES + n1] + acc1[r]);
                outt[(size_t)m * NRES + n1] = v1;
            }
            hnext[m * K_H_PAD + n0] = (_Float16)v0;   // padded cols stay 0
            hnext[m * K_H_PAD + n1] = (_Float16)v1;
        }
        grid_barrier(bar, &phase);                  // h_t visible before t+1
    }
}

// ---------------------------------------------------------------------------
// Host launcher
// ---------------------------------------------------------------------------
extern "C" void kernel_launch(void* const* d_in, const int* in_sizes, int n_in,
                              void* d_out, int out_size, void* d_ws, size_t ws_size,
                              hipStream_t stream) {
    const float* x    = (const float*)d_in[0];  // (T,B,IN)
    const float* Win  = (const float*)d_in[1];  // (N,IN)
    const float* bias = (const float*)d_in[2];  // (N)
    const float* Wlsm = (const float*)d_in[3];  // (N,N)
    float*       out  = (float*)d_out;          // (T,B,N)

    char* ws = (char*)d_ws;
    size_t off = 0;
    unsigned* bar   = (unsigned*)(ws + off); off += 256;
    _Float16* h0    = (_Float16*)(ws + off); off += (size_t)BATCH * K_H_PAD * 2;
    _Float16* h1    = (_Float16*)(ws + off); off += (size_t)BATCH * K_H_PAD * 2;
    _Float16* wlsmT = (_Float16*)(ws + off); off += (size_t)N_PAD * K_H_PAD * 2;
    _Float16* winb  = (_Float16*)(ws + off); off += (size_t)N_PAD * K_IN_PAD * 2;
    _Float16* xb    = (_Float16*)(ws + off); off += (size_t)T_STEPS * BATCH * K_IN_PAD * 2;

    init_kernel<<<256, 256, 0, stream>>>(bar, h0);
    cvt_x_kernel<<<8192, 256, 0, stream>>>(x, xb);
    cvt_win_kernel<<<2816, 256, 0, stream>>>(Win, winb);
    cvt_wlsm_kernel<<<4096, 256, 0, stream>>>(Wlsm, wlsmT);

    // Stage A: input projection + bias -> d_out acts as "pre" buffer
    dim3 gridA(N_PAD / 64, (T_STEPS * BATCH) / 64);   // (16, 1000)
    gemm_pre_kernel<<<gridA, 256, 0, stream>>>(xb, winb, bias, out);

    // Stage B: persistent recurrent sweep
    reservoir_kernel<<<NWG_STEP, 256, 0, stream>>>(wlsmT, out, h0, h1, bar);
}
